// GraphAttention_65833258713447
// MI455X (gfx1250) — matrix operand order repaired and checked by reading, not compiled, not run on hardware
//
#include <hip/hip_runtime.h>
#include <hip/hip_bf16.h>

// ---------------------------------------------------------------------------
// GraphAttention transformer forward for MI455X (gfx1250, wave32, WMMA).
// All GEMM-shaped work runs through v_wmma_f32_16x16x32_f16 (f16 in, f32 acc).
// All matrix fragments are contiguous 16B loads (B operands pre-transposed);
// GEMM wave tile is 32x32 (4 independent WMMAs per K-step, 2 loads/WMMA).
// ---------------------------------------------------------------------------

typedef _Float16 v16h __attribute__((ext_vector_type(16)));
typedef float    v8f  __attribute__((ext_vector_type(8)));
typedef unsigned int u32x4 __attribute__((ext_vector_type(4)));

#define LANE (threadIdx.x & 31)

constexpr int kB    = 4;
constexpr int kN    = 1024;
constexpr int kDIM  = 256;
constexpr int kH    = 4;
constexpr int kDH   = 64;
constexpr int kMLP  = 1024;
constexpr int kDEPTH = 4;

__device__ inline v8f wmma32(v16h a, v16h b, v8f c) {
  return __builtin_amdgcn_wmma_f32_16x16x32_f16(false, a, false, b, (short)0, c,
                                                false, false);
}

// A fragment (16x32 f16, row-major source, row = lane&15).
// ISA layout: lanes 0-15 hold K = {0..7,16..23}, lanes 16-31 K = {8..15,24..31}.
__device__ inline v16h load_a_frag(const _Float16* base, int lda, int k0) {
  int hi = (LANE & 16) ? 8 : 0;
  const _Float16* p = base + (size_t)(LANE & 15) * lda + k0 + hi;
  union { v16h v; u32x4 q[2]; } f;
  f.q[0] = *(const u32x4*)(p);
  f.q[1] = *(const u32x4*)(p + 16);
  return f.v;
}

// B fragment (32x16) from memory laid out [n][k] contiguous in k (B^T).
// col = lane&15, K = {0..7,16..23} (lanes 0-15), {8..15,24..31} (lanes 16-31).
__device__ inline v16h load_b_frag_contig(const _Float16* base, int ld) {
  int ko = (LANE & 16) ? 8 : 0;
  const _Float16* p = base + (size_t)(LANE & 15) * ld + ko;
  union { v16h v; u32x4 q[2]; } f;
  f.q[0] = *(const u32x4*)(p);
  f.q[1] = *(const u32x4*)(p + 16);
  return f.v;
}

// ---------------------------------------------------------------------------
// Generic f16 WMMA GEMM: C[M x Nc] = A[M x K] * Bt^T, Bt stored [Nc x K].
// Block: 256 threads = 8 waves (2 x 4), block tile 64 x 128, wave tile 32 x 32.
// No LDS, no barriers: A and B fragments are contiguous global b128 loads.
// mode 0: outH = (f16)acc
// mode 1: resF += acc + bias ; auxH (xT, [B][DIM][N]) = (f16)resF transposed
// mode 2: outH = (f16)gelu(acc + bias)
// mode 3: outF = acc
// mode 4: outH = (f16)acc ; for cols>=DIM scatter V into auxH (vT [B,H,DH,N])
// ---------------------------------------------------------------------------
__launch_bounds__(256)
__global__ void gemm_wmma_f16(const _Float16* __restrict__ A, int lda, long strideA,
                              const _Float16* __restrict__ Bt, long strideB,
                              int M, int Nc, int K, int mode,
                              _Float16* __restrict__ outH, long strideOutH,
                              float* __restrict__ outF,
                              float* __restrict__ resF,
                              _Float16* __restrict__ auxH,
                              const float* __restrict__ bias) {
  int bz = blockIdx.z;
  const _Float16* Ab = A + (size_t)bz * strideA;
  const _Float16* Bb = Bt + (size_t)bz * strideB;
  int w = threadIdx.x >> 5;
  int row0 = blockIdx.y * 64 + (w & 1) * 32;
  int col0 = blockIdx.x * 128 + (w >> 1) * 32;

  v8f acc[2][2];
#pragma unroll
  for (int i = 0; i < 2; ++i)
#pragma unroll
    for (int j = 0; j < 2; ++j)
      acc[i][j] = (v8f){0.f,0.f,0.f,0.f,0.f,0.f,0.f,0.f};

  const _Float16* a0r = Ab + (size_t)row0 * lda;
  const _Float16* a1r = a0r + (size_t)16 * lda;
  const _Float16* b0r = Bb + (size_t)col0 * K;
  const _Float16* b1r = b0r + (size_t)16 * K;

#pragma unroll 2
  for (int k0 = 0; k0 < K; k0 += 32) {
    v16h a0 = load_a_frag(a0r, lda, k0);
    v16h a1 = load_a_frag(a1r, lda, k0);
    v16h f0 = load_b_frag_contig(b0r + k0, K);
    v16h f1 = load_b_frag_contig(b1r + k0, K);
    acc[0][0] = wmma32(a0, f0, acc[0][0]);
    acc[0][1] = wmma32(a0, f1, acc[0][1]);
    acc[1][0] = wmma32(a1, f0, acc[1][0]);
    acc[1][1] = wmma32(a1, f1, acc[1][1]);
  }

  int lm  = LANE & 15;
  int hi8 = (LANE & 16) ? 8 : 0;
#pragma unroll
  for (int rr = 0; rr < 2; ++rr) {
#pragma unroll
    for (int cc = 0; cc < 2; ++cc) {
#pragma unroll
      for (int r = 0; r < 8; ++r) {
        int gm = row0 + rr * 16 + r + hi8;
        int gn = col0 + cc * 16 + lm;
        size_t idx = (size_t)gm * Nc + gn;
        float v = acc[rr][cc][r];
        if (mode == 0) {
          outH[(size_t)bz * strideOutH + idx] = (_Float16)v;
        } else if (mode == 1) {
          float nv = resF[idx] + v + bias[gn];
          resF[idx] = nv;
          int bI = gm >> 10, j = gm & (kN - 1);
          auxH[((size_t)bI * kDIM + gn) * kN + j] = (_Float16)nv;
        } else if (mode == 2) {
          float t2 = v + bias[gn];
          float g = 0.5f * t2 * (1.0f + erff(t2 * 0.70710678118654752f));
          outH[idx] = (_Float16)g;
        } else if (mode == 3) {
          outF[idx] = v;
        } else {  // mode 4: kv projection, scatter V transposed per head
          _Float16 hv = (_Float16)v;
          outH[idx] = hv;
          if (gn >= kDIM) {
            int hd = gn - kDIM;  // h*64 + d
            int bI = gm >> 10, j = gm & (kN - 1);
            auxH[((size_t)(bI * kH + (hd >> 6)) * kDH + (hd & 63)) * kN + j] = hv;
          }
        }
      }
    }
  }
}

// ---------------------------------------------------------------------------
// Flash attention: one wave per 16 query rows of one (b,h).
// dots = (Q K^T) * scale + bias0 + bias1 ; online softmax ; O = P V.
// K fragments contiguous from kv ([B,N,512]); V fragments contiguous from vT.
// ---------------------------------------------------------------------------
__launch_bounds__(256)
__global__ void flash_attn(const _Float16* __restrict__ qh,   // [B,N,256]
                           const _Float16* __restrict__ kvh,  // [B,N,512] (K|V)
                           const _Float16* __restrict__ vT,   // [B,H,DH,N]
                           const float* __restrict__ bias0,   // [B,H,N,N]
                           const float* __restrict__ bias1,
                           _Float16* __restrict__ oh,         // [B,N,256]
                           float scale) {
  __shared__ __align__(16) _Float16 Pl[8][16 * 32];

  int w = threadIdx.x >> 5;
  int lane = threadIdx.x & 31;
  int b = blockIdx.z, h = blockIdx.y;
  int i0 = blockIdx.x * 128 + w * 16;
  int lm = lane & 15;
  int hi8 = (lane & 16) ? 8 : 0;

  const _Float16* qbase = qh + ((size_t)b * kN + i0) * kDIM + h * kDH;
  v16h qf0 = load_a_frag(qbase, kDIM, 0);
  v16h qf1 = load_a_frag(qbase, kDIM, 32);

  const _Float16* vthead = vT + (size_t)(b * kH + h) * kDH * kN;

  v8f o[4];
#pragma unroll
  for (int c = 0; c < 4; ++c) o[c] = (v8f){0.f,0.f,0.f,0.f,0.f,0.f,0.f,0.f};
  float m[8], l[8];
#pragma unroll
  for (int r = 0; r < 8; ++r) { m[r] = -1e30f; l[r] = 0.f; }

  const float* b0p = bias0 + (size_t)(b * kH + h) * kN * kN;
  const float* b1p = bias1 + (size_t)(b * kH + h) * kN * kN;

  for (int j0 = 0; j0 < kN; j0 += 32) {
    __syncthreads();  // protect previous iteration's Pl reads
    // prefetch next iteration's bias rows (global_prefetch_b8 path)
    if (j0 + 32 < kN) {
      size_t pro = (size_t)(i0 + (lane >> 1)) * kN + j0 + 32;
      __builtin_prefetch(b0p + pro, 0, 0);
      __builtin_prefetch(b1p + pro, 0, 0);
    }
    // dots for two 16-key subtiles (K stored row-major -> contiguous B frags)
    v8f d0 = {0.f,0.f,0.f,0.f,0.f,0.f,0.f,0.f};
    v8f d1 = {0.f,0.f,0.f,0.f,0.f,0.f,0.f,0.f};
    {
      const _Float16* kb0 = kvh + ((size_t)b * kN + j0) * (2 * kDIM) + h * kDH;
      const _Float16* kb1 = kb0 + (size_t)16 * (2 * kDIM);
      d0 = wmma32(qf0, load_b_frag_contig(kb0, 2 * kDIM), d0);
      d0 = wmma32(qf1, load_b_frag_contig(kb0 + 32, 2 * kDIM), d0);
      d1 = wmma32(qf0, load_b_frag_contig(kb1, 2 * kDIM), d1);
      d1 = wmma32(qf1, load_b_frag_contig(kb1 + 32, 2 * kDIM), d1);
    }
    // scale + bias
#pragma unroll
    for (int r = 0; r < 8; ++r) {
      size_t ro = (size_t)(i0 + r + hi8) * kN + j0 + lm;
      d0[r] = d0[r] * scale + b0p[ro] + b1p[ro];
      d1[r] = d1[r] * scale + b0p[ro + 16] + b1p[ro + 16];
    }
    // online softmax (each row lives across 16 lanes of a half-wave)
#pragma unroll
    for (int r = 0; r < 8; ++r) {
      float tm = fmaxf(d0[r], d1[r]);
      for (int s = 1; s < 16; s <<= 1) tm = fmaxf(tm, __shfl_xor(tm, s, 32));
      float mn = fmaxf(m[r], tm);
      float alpha = expf(m[r] - mn);
      float p0 = expf(d0[r] - mn);
      float p1 = expf(d1[r] - mn);
      float rs = p0 + p1;
      for (int s = 1; s < 16; s <<= 1) rs += __shfl_xor(rs, s, 32);
      l[r] = l[r] * alpha + rs;
      m[r] = mn;
      o[0][r] *= alpha; o[1][r] *= alpha; o[2][r] *= alpha; o[3][r] *= alpha;
      Pl[w][(r + hi8) * 32 + lm] = (_Float16)p0;
      Pl[w][(r + hi8) * 32 + 16 + lm] = (_Float16)p1;
    }
    __syncthreads();
    // O += P (16x32) * V (32x64); V fragments contiguous from vT
    v16h pa = load_a_frag(&Pl[w][0], 32, 0);
#pragma unroll
    for (int c = 0; c < 4; ++c) {
      v16h bv = load_b_frag_contig(vthead + (size_t)(c * 16) * kN + j0, kN);
      o[c] = wmma32(pa, bv, o[c]);
    }
  }

#pragma unroll
  for (int r = 0; r < 8; ++r) {
    float inv = 1.0f / l[r];
    int row = i0 + r + hi8;
    _Float16* dst = oh + ((size_t)b * kN + row) * kDIM + h * kDH + lm;
    dst[0]  = (_Float16)(o[0][r] * inv);
    dst[16] = (_Float16)(o[1][r] * inv);
    dst[32] = (_Float16)(o[2][r] * inv);
    dst[48] = (_Float16)(o[3][r] * inv);
  }
}

// ---------------------------------------------------------------------------
// LayerNorm over 256 columns, fp32 in -> f16 out (optional ReLU).
// ---------------------------------------------------------------------------
__launch_bounds__(256)
__global__ void layernorm_f16(const float* __restrict__ in,
                              const float* __restrict__ g,
                              const float* __restrict__ be,
                              _Float16* __restrict__ out, int relu) {
  __shared__ float red[256];
  size_t row = blockIdx.x;
  int c = threadIdx.x;
  float v = in[row * 256 + c];
  red[c] = v;
  __syncthreads();
  for (int s = 128; s > 0; s >>= 1) {
    if (c < s) red[c] += red[c + s];
    __syncthreads();
  }
  float mu = red[0] * (1.0f / 256.0f);
  __syncthreads();
  float d = v - mu;
  red[c] = d * d;
  __syncthreads();
  for (int s = 128; s > 0; s >>= 1) {
    if (c < s) red[c] += red[c + s];
    __syncthreads();
  }
  float var = red[0] * (1.0f / 256.0f);
  float y = d * rsqrtf(var + 1e-5f) * g[c] + be[c];
  if (relu) y = fmaxf(y, 0.f);
  out[row * 256 + c] = (_Float16)y;
}

// M[b,i,j] = mean_h(bias0) * mean_h(bias1)
__global__ void bias_mean_prod(const float* __restrict__ bias,
                               float* __restrict__ Mout) {
  size_t idx = (size_t)blockIdx.x * 256 + threadIdx.x;
  size_t nn = (size_t)kN * kN;
  size_t bI = idx / nn, rem = idx % nn;
  const float* p0 = bias + bI * kH * nn + rem;
  const float* p1 = bias + (size_t)kB * kH * nn + bI * kH * nn + rem;
  float a = 0.f, s = 0.f;
#pragma unroll
  for (int h = 0; h < kH; ++h) { a += p0[(size_t)h * nn]; s += p1[(size_t)h * nn]; }
  Mout[idx] = (a * 0.25f) * (s * 0.25f);
}

// row softmax over 1024 cols, fp32 in -> f16 out
__launch_bounds__(256)
__global__ void softmax_row_f16(const float* __restrict__ in,
                                _Float16* __restrict__ out) {
  __shared__ float red[256];
  size_t row = blockIdx.x;
  int t = threadIdx.x;
  const float* p = in + row * 1024;
  float v[4];
  float mx = -1e30f;
#pragma unroll
  for (int k = 0; k < 4; ++k) { v[k] = p[t + k * 256]; mx = fmaxf(mx, v[k]); }
  red[t] = mx;
  __syncthreads();
  for (int s = 128; s > 0; s >>= 1) {
    if (t < s) red[t] = fmaxf(red[t], red[t + s]);
    __syncthreads();
  }
  mx = red[0];
  __syncthreads();
  float sum = 0.f;
#pragma unroll
  for (int k = 0; k < 4; ++k) { v[k] = expf(v[k] - mx); sum += v[k]; }
  red[t] = sum;
  __syncthreads();
  for (int s = 128; s > 0; s >>= 1) {
    if (t < s) red[t] += red[t + s];
    __syncthreads();
  }
  float inv = 1.0f / red[0];
#pragma unroll
  for (int k = 0; k < 4; ++k)
    out[row * 1024 + t + k * 256] = (_Float16)(v[k] * inv);
}

// Tiled transpose + f32->f16 convert: in [K x Nc] -> out [Nc x K], per layer z.
__global__ void transpose_f32_to_f16(const float* __restrict__ in,
                                     _Float16* __restrict__ out, int K, int Nc) {
  __shared__ _Float16 t[32][33];
  size_t lo = (size_t)blockIdx.z * K * Nc;
  const float* inb = in + lo;
  _Float16* outb = out + lo;
  int n0 = blockIdx.x * 32, k0 = blockIdx.y * 32;
#pragma unroll
  for (int i = 0; i < 4; ++i) {
    int k = k0 + threadIdx.y + i * 8;
    t[threadIdx.y + i * 8][threadIdx.x] =
        (_Float16)inb[(size_t)k * Nc + n0 + threadIdx.x];
  }
  __syncthreads();
#pragma unroll
  for (int i = 0; i < 4; ++i) {
    int n = n0 + threadIdx.y + i * 8;
    outb[(size_t)n * K + k0 + threadIdx.x] = t[threadIdx.x][threadIdx.y + i * 8];
  }
}

// x init: fp32 copy + transposed f16 mirror xT[b][d][j]
__global__ void init_x(const float* __restrict__ xin, float* __restrict__ xf,
                       _Float16* __restrict__ xT, size_t n) {
  size_t i = (size_t)blockIdx.x * 256 + threadIdx.x;
  if (i < n) {
    float v = xin[i];
    xf[i] = v;
    size_t bI = i / ((size_t)kN * kDIM);
    size_t rem = i % ((size_t)kN * kDIM);
    size_t j = rem / kDIM, d = rem % kDIM;
    xT[(bI * kDIM + d) * kN + j] = (_Float16)v;
  }
}

// ---------------------------------------------------------------------------
extern "C" void kernel_launch(void* const* d_in, const int* in_sizes, int n_in,
                              void* d_out, int out_size, void* d_ws, size_t ws_size,
                              hipStream_t stream) {
  (void)in_sizes; (void)n_in; (void)out_size; (void)ws_size;
  const float* x_in  = (const float*)d_in[0];
  const float* bias  = (const float*)d_in[1];
  const float* ln1g  = (const float*)d_in[2];
  const float* ln1b  = (const float*)d_in[3];
  const float* Wkv   = (const float*)d_in[4];
  const float* Wq    = (const float*)d_in[5];
  const float* Wo    = (const float*)d_in[6];
  const float* bo    = (const float*)d_in[7];
  const float* ln2g  = (const float*)d_in[8];
  const float* ln2b  = (const float*)d_in[9];
  const float* W1    = (const float*)d_in[10];
  const float* b1    = (const float*)d_in[11];
  const float* W2    = (const float*)d_in[12];
  const float* b2    = (const float*)d_in[13];
  const float* Wg    = (const float*)d_in[14];
  const float* lngg  = (const float*)d_in[15];
  const float* lngb  = (const float*)d_in[16];

  char* ws = (char*)d_ws;
  size_t off = 0;
  auto alloc = [&](size_t bytes) -> char* {
    off = (off + 255) & ~(size_t)255;
    char* p = ws + off;
    off += bytes;
    return p;
  };

  const size_t nBND = (size_t)kB * kN * kDIM;  // 1M elems
  float*     x_f   = (float*)alloc(nBND * 4);
  _Float16*  xT_h  = (_Float16*)alloc(nBND * 2);      // [B][DIM][N]
  float*     Mbuf  = (float*)alloc((size_t)kB * kN * kN * 4);  // reused: t, hmlp
  _Float16*  G_h   = (_Float16*)alloc((size_t)kB * kN * kN * 2);
  _Float16*  gx_h  = (_Float16*)alloc(nBND * 2);
  _Float16*  xma_h = (_Float16*)alloc(nBND * 2);
  _Float16*  xn_h  = (_Float16*)alloc(nBND * 2);
  _Float16*  kv_h  = (_Float16*)alloc(nBND * 2 * 2);
  _Float16*  vT_h  = (_Float16*)alloc(nBND * 2);      // [B][H][DH][N]
  _Float16*  q_h   = (_Float16*)alloc(nBND * 2);
  _Float16*  o_h   = (_Float16*)alloc(nBND * 2);
  _Float16*  xn2_h = (_Float16*)alloc(nBND * 2);
  _Float16*  WgT   = (_Float16*)alloc((size_t)kDIM * kDIM * 2);
  _Float16*  WkvT  = (_Float16*)alloc((size_t)kDEPTH * kDIM * 2 * kDIM * 2);
  _Float16*  WqT   = (_Float16*)alloc((size_t)kDEPTH * kDIM * kDIM * 2);
  _Float16*  WoT   = (_Float16*)alloc((size_t)kDEPTH * kDIM * kDIM * 2);
  _Float16*  W1T   = (_Float16*)alloc((size_t)kDEPTH * kDIM * kMLP * 2);
  _Float16*  W2T   = (_Float16*)alloc((size_t)kDEPTH * kMLP * kDIM * 2);
  // reuse the M region after G is built
  float*     t_f   = Mbuf;                               // [B*N,256] fp32
  _Float16*  h_h   = (_Float16*)((char*)Mbuf + nBND * 4);  // [B*N,1024] f16

  auto trp = [&](const float* src, _Float16* dst, int K, int Nc, int layers) {
    transpose_f32_to_f16<<<dim3(Nc / 32, K / 32, layers), dim3(32, 8), 0, stream>>>(
        src, dst, K, Nc);
  };
  trp(Wg,  WgT,  kDIM, kDIM, 1);
  trp(Wkv, WkvT, kDIM, 2 * kDIM, kDEPTH);
  trp(Wq,  WqT,  kDIM, kDIM, kDEPTH);
  trp(Wo,  WoT,  kDIM, kDIM, kDEPTH);
  trp(W1,  W1T,  kDIM, kMLP, kDEPTH);
  trp(W2,  W2T,  kMLP, kDIM, kDEPTH);

  init_x<<<dim3((unsigned)((nBND + 255) / 256)), dim3(256), 0, stream>>>(
      x_in, x_f, xT_h, nBND);

  bias_mean_prod<<<dim3((unsigned)((size_t)kB * kN * kN / 256)), dim3(256), 0,
                   stream>>>(bias, Mbuf);
  softmax_row_f16<<<dim3(kB * kN), dim3(256), 0, stream>>>(Mbuf, G_h);

  const float* bias0 = bias;
  const float* bias1 = bias + (size_t)kB * kH * kN * kN;
  const long sNN = (long)kN * kN;
  const long sND = (long)kN * kDIM;
  const int Mtot = kB * kN;

  auto gemm = [&](const _Float16* A, int lda, long sA,
                  const _Float16* Bt, long sB,
                  int M, int Nc, int K, int batch, int mode,
                  _Float16* outH, long sOH, float* outF,
                  float* resF, _Float16* auxH, const float* bb) {
    dim3 grid(Nc / 128, M / 64, batch);
    gemm_wmma_f16<<<grid, dim3(256), 0, stream>>>(A, lda, sA, Bt, sB, M, Nc, K,
                                                  mode, outH, sOH, outF, resF,
                                                  auxH, bb);
  };

  for (int l = 0; l < kDEPTH; ++l) {
    // gx = G @ x   (batched over B; B operand = xT)
    gemm(G_h, kN, sNN, xT_h, sND, kN, kDIM, kN, kB, /*mode*/0,
         gx_h, sND, nullptr, nullptr, nullptr, nullptr);
    // t = gx @ Wg  -> fp32
    gemm(gx_h, kDIM, 0, WgT, 0, Mtot, kDIM, kDIM, 1, /*mode*/3,
         nullptr, 0, t_f, nullptr, nullptr, nullptr);
    // x_ma = relu(LN(t))
    layernorm_f16<<<dim3(Mtot), dim3(256), 0, stream>>>(t_f, lngg, lngb, xma_h, 1);
    // xn = LN(x)
    layernorm_f16<<<dim3(Mtot), dim3(256), 0, stream>>>(x_f, ln1g + l * kDIM,
                                                        ln1b + l * kDIM, xn_h, 0);
    // kv = xn @ Wkv[l]  (also scatters V transposed into vT)
    gemm(xn_h, kDIM, 0, WkvT + (size_t)l * kDIM * 2 * kDIM, 0,
         Mtot, 2 * kDIM, kDIM, 1, /*mode*/4, kv_h, 0, nullptr, nullptr, vT_h,
         nullptr);
    // q = x_ma @ Wq[l]
    gemm(xma_h, kDIM, 0, WqT + (size_t)l * kDIM * kDIM, 0,
         Mtot, kDIM, kDIM, 1, 0, q_h, 0, nullptr, nullptr, nullptr, nullptr);
    // attention
    flash_attn<<<dim3(kN / 128, kH, kB), dim3(256), 0, stream>>>(
        q_h, kv_h, vT_h, bias0, bias1, o_h, 0.0625f /* DIM^-0.5 */);
    // x = x + o @ Wo[l] + bo[l]   (updates x_f and xT mirror)
    gemm(o_h, kDIM, 0, WoT + (size_t)l * kDIM * kDIM, 0,
         Mtot, kDIM, kDIM, 1, 1, nullptr, 0, nullptr, x_f, xT_h, bo + l * kDIM);
    // xn2 = LN(x)
    layernorm_f16<<<dim3(Mtot), dim3(256), 0, stream>>>(x_f, ln2g + l * kDIM,
                                                        ln2b + l * kDIM, xn2_h, 0);
    // h = gelu(xn2 @ W1[l] + b1[l])
    gemm(xn2_h, kDIM, 0, W1T + (size_t)l * kDIM * kMLP, 0,
         Mtot, kMLP, kDIM, 1, 2, h_h, 0, nullptr, nullptr, nullptr, b1 + l * kMLP);
    // x = x + h @ W2[l] + b2[l]
    gemm(h_h, kMLP, 0, W2T + (size_t)l * kMLP * kDIM, 0,
         Mtot, kDIM, kMLP, 1, 1, nullptr, 0, nullptr, x_f, xT_h, b2 + l * kDIM);
  }

  hipMemcpyAsync(d_out, x_f, nBND * sizeof(float), hipMemcpyDeviceToDevice, stream);
}
